// MonotonicAttention_41489384079889
// MI455X (gfx1250) — compile-verified, matrix operand order checked by reading, not run on hardware
//
#include <hip/hip_runtime.h>
#include <hip/hip_bf16.h>
#include <math.h>

// Problem constants (match reference)
#define BB   8
#define TQd  1024
#define TKd  256
#define EE   512
#define HH   8
#define HDd  64
#define MIN_LP   (-10000.0f)
#define CLAMPV   3.8918202981106265f   // -log(1/0.98 - 1)

typedef __attribute__((ext_vector_type(16))) _Float16 v16h;
typedef __attribute__((ext_vector_type(8)))  _Float16 v8h;
typedef __attribute__((ext_vector_type(8)))  float    v8f;
typedef __attribute__((ext_vector_type(4)))  int      v4i;

// CDNA5 async global->LDS staging (ASYNCcnt path); pointer params are
// generic-AS `v4i*` on this toolchain (probe-confirmed). Guarded fallback.
#if __has_builtin(__builtin_amdgcn_global_load_async_to_lds_b128) && \
    __has_builtin(__builtin_amdgcn_s_wait_asynccnt)
#define USE_ASYNC_LDS 1
#else
#define USE_ASYNC_LDS 0
#endif

// ---------------------------------------------------------------------------
// Generic batched WMMA GEMM:  C[z] = A[z] (MxK, row-major, lda) * B[z] (KxN)
// B is addressed as element(n,k) = Bm[n*ldbn + k]  (column-n view, contiguous
// in k) — weights, xk, and pre-transposed xv all satisfy this.
// Batch z decomposes as b=z>>3, h=z&7 with separate strides (attention heads).
// Block tile 128x64, BK=32; 8 waves, each owns a 2x2 set of 16x16 tiles.
// Double-buffered LDS with async-to-LDS loads overlapping WMMA; last chunk
// peeled so the inner loop is branch-free.
// mode: 0 = f16 out + bias, 1 = f32 out + bias, 2 = f32 tanh-clamp scores,
//       3 = f16 out, no bias
// ---------------------------------------------------------------------------
struct GemmP {
  const _Float16* A;  long long sAb, sAh; int lda;
  const _Float16* Bm; long long sBb, sBh; int ldbn;
  const float*    bias;
  void*           C;  long long sCb, sCh; int ldc;
  int M, N, K;
  int mode;
};

// Load one 16x32 f16 fragment (A-style or column-major-B-style) from LDS.
// Per ISA 16-bit layout: lane L holds row (rowBase + L%16); halves j=0..15 map
// to K = (j<8 ? 0 : 16) + (L<16 ? 0 : 8) + (j%8)  -> two contiguous 8-half runs.
__device__ inline v16h load_frag16(const _Float16* lds, int rowBase) {
  const int lane = threadIdx.x & 31;
  const int row  = rowBase + (lane & 15);
  const int klo  = (lane & 16) ? 8 : 0;
  const _Float16* p = lds + row * 32 + klo;
  v8h lo = *(const v8h*)(p);
  v8h hi = *(const v8h*)(p + 16);
  v16h r;
#pragma unroll
  for (int i = 0; i < 8; ++i) { r[i] = lo[i]; r[i + 8] = hi[i]; }
  return r;
}

__device__ inline void mma_chunk(const _Float16* sA, const _Float16* sB,
                                 int mt0, int nt0,
                                 v8f& acc00, v8f& acc01, v8f& acc10, v8f& acc11) {
  v16h a0 = load_frag16(sA, mt0 * 16);
  v16h a1 = load_frag16(sA, (mt0 + 1) * 16);
  v16h b0 = load_frag16(sB, nt0 * 16);
  v16h b1 = load_frag16(sB, (nt0 + 1) * 16);
  acc00 = __builtin_amdgcn_wmma_f32_16x16x32_f16(false, a0, false, b0,
                                                 (short)0, acc00, false, false);
  acc01 = __builtin_amdgcn_wmma_f32_16x16x32_f16(false, a0, false, b1,
                                                 (short)0, acc01, false, false);
  acc10 = __builtin_amdgcn_wmma_f32_16x16x32_f16(false, a1, false, b0,
                                                 (short)0, acc10, false, false);
  acc11 = __builtin_amdgcn_wmma_f32_16x16x32_f16(false, a1, false, b1,
                                                 (short)0, acc11, false, false);
}

__global__ __launch_bounds__(256) void gemm_wmma_f16(GemmP p) {
  __shared__ __align__(16) _Float16 sA[2][128 * 32];  // [m][k], double buffered
  __shared__ __align__(16) _Float16 sB[2][64 * 32];   // [n][k] (column-major B)

  const int tid  = threadIdx.x;
  const int wave = tid >> 5;
  const int lane = tid & 31;
  // 8x4 grid of 16x16 tiles; wave w owns rows {2rg,2rg+1}, cols {2cg,2cg+1}
  const int mt0 = (wave >> 1) << 1;
  const int nt0 = (wave & 1) << 1;

  const int z = blockIdx.z;
  const int b = z >> 3, h = z & 7;
  const _Float16* Ab = p.A  + (long long)b * p.sAb + (long long)h * p.sAh;
  const _Float16* Bb = p.Bm + (long long)b * p.sBb + (long long)h * p.sBh;
  const long long cOff = (long long)b * p.sCb + (long long)h * p.sCh;

  const int row0 = blockIdx.y * 128;
  const int col0 = blockIdx.x * 64;

  v8f acc00 = {}, acc01 = {}, acc10 = {}, acc11 = {};

  // Staging assignments (all 16-byte aligned, contiguous in k):
  // A tile: 128x32 halves; thread t copies row (t>>1), cols [(t&1)*16, +16)
  const int aOff = (tid >> 1) * 32 + ((tid & 1) << 4);
  // B tile: 64x32 halves; thread t copies row (t>>2), cols [(t&3)*8, +8)
  const int bOff = (tid >> 2) * 32 + ((tid & 3) << 3);

  const _Float16* as = Ab + (long long)(row0 + (tid >> 1)) * p.lda + ((tid & 1) << 4);
  const _Float16* bs = Bb + (long long)(col0 + (tid >> 2)) * p.ldbn + ((tid & 3) << 3);

#if USE_ASYNC_LDS
  const int nk = p.K >> 5;            // always >= 2 in this pipeline
  // Prologue: chunk 0 into buffer 0.
  __builtin_amdgcn_global_load_async_to_lds_b128((v4i*)as,       (v4i*)(sA[0] + aOff),     0, 0);
  __builtin_amdgcn_global_load_async_to_lds_b128((v4i*)(as + 8), (v4i*)(sA[0] + aOff + 8), 0, 0);
  __builtin_amdgcn_global_load_async_to_lds_b128((v4i*)bs,       (v4i*)(sB[0] + bOff),     0, 0);
  as += 32; bs += 32;

  // Branch-free steady state: always issue the next chunk into the other
  // buffer, drain only the 3 older loads, compute on the current buffer.
  for (int i = 0; i < nk - 1; ++i) {
    const int cur = i & 1;
    const int nxt = cur ^ 1;
    __builtin_amdgcn_global_load_async_to_lds_b128((v4i*)as,       (v4i*)(sA[nxt] + aOff),     0, 0);
    __builtin_amdgcn_global_load_async_to_lds_b128((v4i*)(as + 8), (v4i*)(sA[nxt] + aOff + 8), 0, 0);
    __builtin_amdgcn_global_load_async_to_lds_b128((v4i*)bs,       (v4i*)(sB[nxt] + bOff),     0, 0);
    as += 32; bs += 32;
    __builtin_amdgcn_s_wait_asynccnt(3);
    __syncthreads();   // all waves' chunk-i data visible
    mma_chunk(sA[cur], sB[cur], mt0, nt0, acc00, acc01, acc10, acc11);
    __syncthreads();   // reads of buf[cur] done before it is refilled
  }
  // Peeled final chunk.
  __builtin_amdgcn_s_wait_asynccnt(0);
  __syncthreads();
  mma_chunk(sA[(nk - 1) & 1], sB[(nk - 1) & 1], mt0, nt0, acc00, acc01, acc10, acc11);
#else
  for (int k0 = 0; k0 < p.K; k0 += 32, as += 32, bs += 32) {
    *(v8h*)(sA[0] + aOff)     = *(const v8h*)(as);
    *(v8h*)(sA[0] + aOff + 8) = *(const v8h*)(as + 8);
    *(v8h*)(sB[0] + bOff)     = *(const v8h*)(bs);
    if (k0 + 32 < p.K) {  // hint next K-chunk (global_prefetch_b8)
      __builtin_prefetch(as + 32, 0, 1);
      __builtin_prefetch(bs + 32, 0, 1);
    }
    __syncthreads();
    mma_chunk(sA[0], sB[0], mt0, nt0, acc00, acc01, acc10, acc11);
    __syncthreads();
  }
#endif

  // Epilogue. C layout: lane -> N = lane%16, VGPR r -> M = r + (lane>=16 ? 8:0)
  const int laneM = (lane >> 4) << 3;
  const int laneN = lane & 15;
#pragma unroll
  for (int i = 0; i < 2; ++i) {
    const int mBase = row0 + (mt0 + i) * 16 + laneM;
#pragma unroll
    for (int j = 0; j < 2; ++j) {
      const int n = col0 + (nt0 + j) * 16 + laneN;
      v8f acc = (i == 0) ? (j == 0 ? acc00 : acc01) : (j == 0 ? acc10 : acc11);
#pragma unroll
      for (int r = 0; r < 8; ++r) {
        const long long o = cOff + (long long)(mBase + r) * p.ldc + n;
        if (p.mode == 0) {
          ((_Float16*)p.C)[o] = (_Float16)(acc[r] + p.bias[n]);
        } else if (p.mode == 1) {
          ((float*)p.C)[o] = acc[r] + p.bias[n];
        } else if (p.mode == 2) {
          ((float*)p.C)[o] = tanhf(acc[r] * 0.125f) * CLAMPV;
        } else {
          ((_Float16*)p.C)[o] = (_Float16)acc[r];
        }
      }
    }
  }
}

// ---------------------------------------------------------------------------
// f32 -> f16 conversion
// ---------------------------------------------------------------------------
__global__ void cvt_f32_f16(const float* __restrict__ s,
                            _Float16* __restrict__ d, int n) {
  int i = blockIdx.x * 256 + threadIdx.x;
  if (i < n) d[i] = (_Float16)s[i];
}

// ---------------------------------------------------------------------------
// Per-head transpose of xv: (B,TK,E) -> xvT (B,H,HD,TK) so the PV GEMM's
// B view is contiguous in k.
// ---------------------------------------------------------------------------
__global__ void transpose_xv(const _Float16* __restrict__ xv,
                             _Float16* __restrict__ xvT) {
  long long i = (long long)blockIdx.x * 256 + threadIdx.x;  // over B*H*HD*TK
  const int k  = (int)(i & (TKd - 1));
  long long r  = i >> 8;              // B*H*HD
  const int d  = (int)(r & (HDd - 1));
  long long bh = r >> 6;              // b*H + h
  const int h  = (int)(bh & 7);
  const int b  = (int)(bh >> 3);
  xvT[i] = xv[((long long)(b * TKd + k)) * EE + h * HDd + d];
}

// ---------------------------------------------------------------------------
// Monotonic (CTC-style) forward scan. One block per n = b*H + h.
// phi state (TK+1) lives in LDS; 1024 barrier-separated steps over TQ.
// Writes P[n,t,k] = exp(phi_t[k+1] + logsig(-logits[n,t,k])) as f16.
// ---------------------------------------------------------------------------
__device__ inline float logsigf(float x) {
  float ax = fabsf(x);
  return fminf(x, 0.0f) - log1pf(__expf(-ax));
}

__global__ __launch_bounds__(256) void monotonic_scan(
    const float* __restrict__ logits, _Float16* __restrict__ P) {
  const int n = blockIdx.x;        // 0..63
  const int k = threadIdx.x;       // 0..255
  __shared__ float phi[TKd + 1];
  __shared__ float row[TKd];

  phi[k] = (k == 1) ? 0.0f : MIN_LP;
  if (k == 0) phi[TKd] = MIN_LP;
  __syncthreads();

  const float* Ln = logits + (long long)n * TQd * TKd;
  _Float16*    Pn = P      + (long long)n * TQd * TKd;

  for (int t = 0; t < TQd; ++t) {
    const float l = Ln[(long long)t * TKd + k];
    row[k] = l;
    const float pk1 = phi[k + 1];
    const float pk  = phi[k];
    // marginal prob for this (t,k): uses phi BEFORE applying row t
    const float lp = pk1 + logsigf(-l);
    Pn[(long long)t * TKd + k] = (_Float16)__expf(lp);
    __syncthreads();                      // row[] visible; phi reads done
    // phi_{t+1}[k+1] = logaddexp(phi_t[k+1]+logsig(l_k),
    //                            phi_t[k]  +logsig(-lpad_k)), lpad_0 = 0
    const float lprev = (k == 0) ? 0.0f : row[k - 1];
    const float a = pk1 + logsigf(l);
    const float c = pk  + logsigf(-lprev);
    const float m = fmaxf(a, c);
    phi[k + 1] = m + __logf(__expf(a - m) + __expf(c - m));
    __syncthreads();                      // phi update complete
  }
}

// ---------------------------------------------------------------------------
// Launch
// ---------------------------------------------------------------------------
extern "C" void kernel_launch(void* const* d_in, const int* in_sizes, int n_in,
                              void* d_out, int out_size, void* d_ws, size_t ws_size,
                              hipStream_t stream) {
  const float* q    = (const float*)d_in[0];
  const float* kk   = (const float*)d_in[1];
  const float* v    = (const float*)d_in[2];
  const float* wqkv = (const float*)d_in[3];
  const float* bqkv = (const float*)d_in[4];
  const float* wout = (const float*)d_in[5];
  const float* bout = (const float*)d_in[6];

  char* ws = (char*)d_ws;
  size_t off = 0;
  auto alloc = [&](size_t bytes) -> void* {
    void* p = ws + off;
    off += (bytes + 255) & ~(size_t)255;
    return p;
  };

  const int nQ = BB * TQd * EE, nK = BB * TKd * EE;
  _Float16* q16  = (_Float16*)alloc((size_t)nQ * 2);
  _Float16* k16  = (_Float16*)alloc((size_t)nK * 2);
  _Float16* v16  = (_Float16*)alloc((size_t)nK * 2);
  _Float16* w16  = (_Float16*)alloc((size_t)3 * EE * EE * 2);
  _Float16* wo16 = (_Float16*)alloc((size_t)EE * EE * 2);
  _Float16* xq   = (_Float16*)alloc((size_t)nQ * 2);
  _Float16* xk   = (_Float16*)alloc((size_t)nK * 2);
  _Float16* xv   = (_Float16*)alloc((size_t)nK * 2);
  _Float16* xvT  = (_Float16*)alloc((size_t)nK * 2);
  float*    lg   = (float*)   alloc((size_t)BB * HH * TQd * TKd * 4);  // 64 MB
  _Float16* P    = (_Float16*)alloc((size_t)BB * HH * TQd * TKd * 2);  // 32 MB
  _Float16* ao   = (_Float16*)alloc((size_t)nQ * 2);

  // 0) convert inputs + weights to f16
  cvt_f32_f16<<<(nQ + 255) / 256, 256, 0, stream>>>(q,    q16,  nQ);
  cvt_f32_f16<<<(nK + 255) / 256, 256, 0, stream>>>(kk,   k16,  nK);
  cvt_f32_f16<<<(nK + 255) / 256, 256, 0, stream>>>(v,    v16,  nK);
  cvt_f32_f16<<<(3 * EE * EE + 255) / 256, 256, 0, stream>>>(wqkv, w16, 3 * EE * EE);
  cvt_f32_f16<<<(EE * EE + 255) / 256, 256, 0, stream>>>(wout, wo16, EE * EE);

  GemmP p;
  // 1) xq = q @ wq^T + bq   (8192 x 512 x 512)
  p = {q16, 0, 0, EE,  w16, 0, 0, EE,  bqkv,
       xq, 0, 0, EE,  BB * TQd, EE, EE, 0};
  gemm_wmma_f16<<<dim3(EE / 64, (BB * TQd) / 128, 1), 256, 0, stream>>>(p);
  // 2) xk
  p = {k16, 0, 0, EE,  w16 + (size_t)EE * EE, 0, 0, EE,  bqkv + EE,
       xk, 0, 0, EE,  BB * TKd, EE, EE, 0};
  gemm_wmma_f16<<<dim3(EE / 64, (BB * TKd) / 128, 1), 256, 0, stream>>>(p);
  // 3) xv
  p = {v16, 0, 0, EE,  w16 + (size_t)2 * EE * EE, 0, 0, EE,  bqkv + 2 * EE,
       xv, 0, 0, EE,  BB * TKd, EE, EE, 0};
  gemm_wmma_f16<<<dim3(EE / 64, (BB * TKd) / 128, 1), 256, 0, stream>>>(p);

  // 3b) per-head transpose of xv for contiguous-k B staging in PV GEMM
  transpose_xv<<<(BB * HH * HDd * TKd) / 256, 256, 0, stream>>>(xv, xvT);

  // 4) scores: per (b,h)  xq_h (1024x64) @ xk_h^T (64x256) -> tanh-clamped f32
  p = {xq, (long long)TQd * EE, HDd, EE,
       xk, (long long)TKd * EE, HDd, EE,
       nullptr,
       lg, (long long)HH * TQd * TKd, (long long)TQd * TKd, TKd,
       TQd, TKd, HDd, 2};
  gemm_wmma_f16<<<dim3(TKd / 64, TQd / 128, BB * HH), 256, 0, stream>>>(p);

  // 5) monotonic scan -> P = exp(log marginal probs), f16
  monotonic_scan<<<BB * HH, TKd, 0, stream>>>(lg, P);

  // 6) out_h = P (1024x256) @ xvT_h^T (256x64), f16, no bias
  p = {P,   (long long)HH * TQd * TKd, (long long)TQd * TKd, TKd,
       xvT, (long long)HH * HDd * TKd, (long long)HDd * TKd, TKd,
       nullptr,
       ao, (long long)TQd * EE, HDd, EE,
       TQd, HDd, TKd, 3};
  gemm_wmma_f16<<<dim3(HDd / 64, TQd / 128, BB * HH), 256, 0, stream>>>(p);

  // 7) out = ao @ wout^T + bout, f32
  p = {ao, 0, 0, EE,  wo16, 0, 0, EE,  bout,
       d_out, 0, 0, EE,  BB * TQd, EE, EE, 1};
  gemm_wmma_f16<<<dim3(EE / 64, (BB * TQd) / 128, 1), 256, 0, stream>>>(p);

  (void)in_sizes; (void)n_in; (void)out_size; (void)ws_size;
}
